// HDRNetColorModel_9405978378862
// MI455X (gfx1250) — matrix operand order repaired and checked by reading, not compiled, not run on hardware
//
#include <hip/hip_runtime.h>
#include <hip/hip_bf16.h>

typedef __attribute__((ext_vector_type(16))) _Float16 v16h;
typedef __attribute__((ext_vector_type(8)))  _Float16 v8h;
typedef __attribute__((ext_vector_type(8)))  float    v8f;
typedef __attribute__((ext_vector_type(4)))  unsigned int v4u;

union F16Frag { v16h h; v4u u[2]; };
union H8Pack  { v8h h; v4u u; };

// ---------------------------------------------------------------------------
// Stage 1: bilinear downsample x (B,3,1024,1024 f32, NCHW) -> act0 (B,256,256,4 f16)
// channel-last, 4th channel zero-padded so each pixel is one aligned 8-byte slot.
// ---------------------------------------------------------------------------
__global__ void downsample_kernel(const float* __restrict__ x,
                                  _Float16* __restrict__ act0, int total) {
  int id = blockIdx.x * blockDim.x + threadIdx.x;
  if (id >= total) return;                 // total = B*65536
  int b = id >> 16;
  int gp = id & 65535;
  int oy = gp >> 8, ox = gp & 255;
  float ys = fminf(fmaxf((oy + 0.5f) * 4.f - 0.5f, 0.f), 1023.f);
  float xs = fminf(fmaxf((ox + 0.5f) * 4.f - 0.5f, 0.f), 1023.f);
  int y0 = (int)ys, x0 = (int)xs;
  float wy = ys - (float)y0, wx = xs - (float)x0;
  int y1 = (y0 + 1 > 1023) ? 1023 : y0 + 1;
  int x1 = (x0 + 1 > 1023) ? 1023 : x0 + 1;
  const float* xb = x + (size_t)b * 3u * 1048576u;
  _Float16* o = act0 + (size_t)id * 4;
#pragma unroll
  for (int c = 0; c < 3; ++c) {
    const float* pc = xb + (size_t)c * 1048576u;
    float p00 = pc[y0 * 1024 + x0], p01 = pc[y0 * 1024 + x1];
    float p10 = pc[y1 * 1024 + x0], p11 = pc[y1 * 1024 + x1];
    float v = (1.f - wy) * ((1.f - wx) * p00 + wx * p01) +
              wy * ((1.f - wx) * p10 + wx * p11);
    o[c] = (_Float16)v;
  }
  o[3] = (_Float16)0.f;
}

// ---------------------------------------------------------------------------
// Stage 2: conv1 3->16 (K=27, too small for WMMA) — plain VALU, one thread/pixel.
// ---------------------------------------------------------------------------
__global__ void conv1_kernel(const _Float16* __restrict__ in,   // [B][256][256][4]
                             const float* __restrict__ w,       // (16,3,3,3)
                             const float* __restrict__ bias,    // (16)
                             _Float16* __restrict__ out,        // [B][256][256][16]
                             int total) {
  int id = blockIdx.x * blockDim.x + threadIdx.x;
  if (id >= total) return;                 // total = B*65536
  int b = id >> 16;
  int gp = id & 65535;
  int y = gp >> 8, x = gp & 255;
  float acc[16];
#pragma unroll
  for (int c = 0; c < 16; ++c) acc[c] = bias[c];
  const _Float16* ib = in + (size_t)b * 65536u * 4u;
#pragma unroll
  for (int t = 0; t < 9; ++t) {
    int iy = y + t / 3 - 1;
    int ix = x + t % 3 - 1;
    if (iy < 0 || iy > 255 || ix < 0 || ix > 255) continue;
    const _Float16* p = ib + (size_t)(iy * 256 + ix) * 4;
    float i0 = (float)p[0], i1 = (float)p[1], i2 = (float)p[2];
#pragma unroll
    for (int c = 0; c < 16; ++c) {
      const float* wc = w + c * 27 + t;   // w[c][ci][tap] = w[c*27 + ci*9 + t]
      acc[c] += wc[0] * i0 + wc[9] * i1 + wc[18] * i2;
    }
  }
  H8Pack r0, r1;
#pragma unroll
  for (int c = 0; c < 8; ++c) r0.h[c] = (_Float16)fmaxf(acc[c], 0.f);
#pragma unroll
  for (int c = 0; c < 8; ++c) r1.h[c] = (_Float16)fmaxf(acc[8 + c], 0.f);
  _Float16* op = out + (size_t)id * 16;
  *(v4u*)(op) = r0.u;
  *(v4u*)(op + 8) = r1.u;
}

// ---------------------------------------------------------------------------
// Weight prep: f32 (COUT,CIN,3,3) -> f16 [COUT][KP] im2col layout, K = tap*CIN+ci,
// zero-padded to KP (multiple of 32).
// ---------------------------------------------------------------------------
__global__ void wprep_kernel(const float* __restrict__ w, _Float16* __restrict__ wp,
                             int CIN, int KP, int total) {
  int i = blockIdx.x * blockDim.x + threadIdx.x;
  if (i >= total) return;                  // total = COUT*KP
  int co = i / KP, k = i % KP;
  float v = 0.f;
  int K9 = CIN * 9;
  if (k < K9) {
    int t = k / CIN, ci = k % CIN;
    v = w[(co * CIN + ci) * 9 + t];
  }
  wp[i] = (_Float16)v;
}

// ---------------------------------------------------------------------------
// Core: implicit-GEMM 3x3 conv via v_wmma_f32_16x16x32_f16.
//   in : [B][256][256][CIN]  f16 channel-last
//   wp : [COUT][KP]          f16 im2col weights
//   out: [B][256][256][COUT] f16
// Block = 256 threads (8 waves), 32 pixels per block (one row segment).
// LDS holds the 32 x KP im2col tile (B-operand, K-major per pixel).
// Each wave computes 16x16 output tiles: A frag from global weights,
// B frag from LDS, K-loop of KP/32 WMMAs, b128 packed f16 epilogue.
// ---------------------------------------------------------------------------
template <int CIN, int COUT, int KP, bool RELU>
__global__ __launch_bounds__(256) void conv_wmma_kernel(
    const _Float16* __restrict__ in, const _Float16* __restrict__ wp,
    const float* __restrict__ bias, _Float16* __restrict__ out) {
  __shared__ __align__(16) _Float16 sB[32 * KP];
  const int b = blockIdx.z;
  const int gp0 = blockIdx.x * 32;         // 32 pixels, same row (32 | 256)
  const int y = gp0 >> 8;
  const int xbase = gp0 & 255;

  // ---- im2col stage to LDS (dword granular, coalesced over channels) ----
  const int dwPix = CIN / 2;               // dwords per (pixel,tap) segment
  const unsigned int* inu =
      (const unsigned int*)(in + (size_t)b * 65536u * (unsigned)CIN);
  unsigned int* su = (unsigned int*)sB;
  const int totalDw = 32 * 9 * dwPix;
  for (int i = threadIdx.x; i < totalDw; i += 256) {
    int dw = i % dwPix;
    int tp = i / dwPix;
    int t = tp % 9;
    int p = tp / 9;
    int iy = y + t / 3 - 1;
    int ix = xbase + p + t % 3 - 1;
    unsigned int val = 0u;
    if (iy >= 0 && iy < 256 && ix >= 0 && ix < 256)
      val = inu[(size_t)(iy * 256 + ix) * dwPix + dw];
    su[p * (KP / 2) + t * dwPix + dw] = val;
  }
  // zero the K padding columns (KP - 9*CIN)
  const int K9 = CIN * 9;
  for (int i = threadIdx.x; i < 32 * (KP - K9); i += 256) {
    int p = i / (KP - K9);
    int k = K9 + i % (KP - K9);
    sB[p * KP + k] = (_Float16)0.f;
  }
  __syncthreads();

  // ---- WMMA compute ----
  const int wave = threadIdx.x >> 5;
  const int lane = threadIdx.x & 31;
  const int ln = lane & 15;
  const int hs = lane >> 4;                // K-half select per ISA layout
  const int nTiles = (COUT / 16) * 2;      // cout-tiles x 2 pixel-tiles
  for (int tile = wave; tile < nTiles; tile += 8) {
    int ct = tile >> 1, pt = tile & 1;
    v8f acc = {};
    const _Float16* wr = wp + (size_t)(ct * 16 + ln) * KP;
    const _Float16* br = sB + (size_t)(pt * 16 + ln) * KP + hs * 16;
    for (int kk = 0; kk < KP; kk += 32) {
      F16Frag a, bm;
      // A 16x32 f16: lane row = ln; VGPR0-3 hold K = kk + hs*16 .. +7,
      //              VGPR4-7 hold K = kk + hs*16 + 8 .. +7
      a.u[0] = *(const v4u*)(wr + kk + hs * 16);
      a.u[1] = *(const v4u*)(wr + kk + hs * 16 + 8);
      // B 32x16 f16: lane column = ln (pixel), K range = kk + hs*16 .. +15
      bm.u[0] = *(const v4u*)(br + kk);
      bm.u[1] = *(const v4u*)(br + kk + 8);
      acc = __builtin_amdgcn_wmma_f32_16x16x32_f16(
          false, a.h, false, bm.h, (short)0, acc, false, false);
    }
    // C/D layout: VGPR r -> M = hs*8 + r (consecutive output channels), N = ln.
    int gp = gp0 + pt * 16 + ln;
    int cobase = ct * 16 + hs * 8;
    H8Pack res;
#pragma unroll
    for (int r = 0; r < 8; ++r) {
      float v = acc[r] + bias[cobase + r];
      if (RELU) v = fmaxf(v, 0.f);
      res.h[r] = (_Float16)v;
    }
    *(v4u*)(out + ((size_t)b * 65536u + (unsigned)gp) * (unsigned)COUT + cobase) = res.u;
  }
}

// ---------------------------------------------------------------------------
// Stage 4: 256x256x96 -> bilateral grid (B,12,16,16,8) f32.
// Bilinear 256->16 is exactly a 2x2 average at (16g+7, 16g+8).
// ---------------------------------------------------------------------------
__global__ void grid_build_kernel(const _Float16* __restrict__ act6,
                                  float* __restrict__ grid, int total) {
  int id = blockIdx.x * blockDim.x + threadIdx.x;
  if (id >= total) return;                 // total = B*96*256
  int gx = id & 15;
  int gy = (id >> 4) & 15;
  int c = (id >> 8) % 96;
  int b = (id >> 8) / 96;
  int y0 = gy * 16 + 7, x0 = gx * 16 + 7;
  const _Float16* a = act6 + (size_t)b * 65536u * 96u;
  float v = 0.25f * ((float)a[(size_t)(y0 * 256 + x0) * 96 + c] +
                     (float)a[(size_t)(y0 * 256 + x0 + 1) * 96 + c] +
                     (float)a[(size_t)((y0 + 1) * 256 + x0) * 96 + c] +
                     (float)a[(size_t)((y0 + 1) * 256 + x0 + 1) * 96 + c]);
  int p = c >> 3, d = c & 7;               // c = p*8 + d
  grid[(((size_t)b * 12 + p) * 256 + gy * 16 + gx) * 8 + d] = v;
}

// ---------------------------------------------------------------------------
// Stage 5: fused guide + trilinear slice + affine apply + clip.
// One thread per full-res pixel; grid (96KB/B) stays in L2.
// ---------------------------------------------------------------------------
__global__ void slice_apply_kernel(const float* __restrict__ x,
                                   const float* __restrict__ grid,
                                   float* __restrict__ out, int total) {
  int id = blockIdx.x * blockDim.x + threadIdx.x;
  if (id >= total) return;                 // total = B*1048576
  int b = id >> 20;
  int gp = id & 1048575;
  int y = gp >> 10, xc = gp & 1023;
  const float* xb = x + (size_t)b * 3u * 1048576u;
  float r = xb[gp], g = xb[1048576u + gp], bl = xb[2u * 1048576u + gp];
  float gd = fminf(fmaxf(0.299f * r + 0.587f * g + 0.114f * bl, 0.f), 1.f);
  float ys = y * (15.f / 1023.f), xs = xc * (15.f / 1023.f);
  int y0 = (int)ys, x0 = (int)xs;
  float wy = ys - (float)y0, wx = xs - (float)x0;
  int y1 = (y0 + 1 > 15) ? 15 : y0 + 1;
  int x1 = (x0 + 1 > 15) ? 15 : x0 + 1;
  float dd = gd * 7.f;
  int d0 = (int)dd;
  if (d0 > 7) d0 = 7;
  int d1 = (d0 + 1 > 7) ? 7 : d0 + 1;
  float wd = fminf(fmaxf(dd - (float)d0, 0.f), 1.f);
  float w00 = (1.f - wy) * (1.f - wx), w01 = (1.f - wy) * wx;
  float w10 = wy * (1.f - wx), w11 = wy * wx;
  int i00 = (y0 * 16 + x0) * 8, i01 = (y0 * 16 + x1) * 8;
  int i10 = (y1 * 16 + x0) * 8, i11 = (y1 * 16 + x1) * 8;
  const float* gb = grid + (size_t)b * 12u * 2048u;
  float sc[12];
#pragma unroll
  for (int i = 0; i < 12; ++i) {
    const float* gi = gb + (size_t)i * 2048u;
    float c00 = gi[i00 + d0] + wd * (gi[i00 + d1] - gi[i00 + d0]);
    float c01 = gi[i01 + d0] + wd * (gi[i01 + d1] - gi[i01 + d0]);
    float c10 = gi[i10 + d0] + wd * (gi[i10 + d1] - gi[i10 + d0]);
    float c11 = gi[i11 + d0] + wd * (gi[i11 + d1] - gi[i11 + d0]);
    sc[i] = w00 * c00 + w01 * c01 + w10 * c10 + w11 * c11;
  }
#pragma unroll
  for (int row = 0; row < 3; ++row) {
    float v = sc[row * 4 + 0] * r + sc[row * 4 + 1] * g + sc[row * 4 + 2] * bl +
              sc[row * 4 + 3];
    v = fminf(fmaxf(v, 0.f), 1.f);
    out[((size_t)(b * 3 + row) << 20) + gp] = v;
  }
}

// ---------------------------------------------------------------------------
extern "C" void kernel_launch(void* const* d_in, const int* in_sizes, int n_in,
                              void* d_out, int out_size, void* d_ws, size_t ws_size,
                              hipStream_t stream) {
  const float* x  = (const float*)d_in[0];
  const float* w1 = (const float*)d_in[1];
  const float* b1 = (const float*)d_in[2];
  const float* w2 = (const float*)d_in[3];
  const float* b2 = (const float*)d_in[4];
  const float* w3 = (const float*)d_in[5];
  const float* b3 = (const float*)d_in[6];
  const float* w4 = (const float*)d_in[7];
  const float* b4 = (const float*)d_in[8];
  const float* w5 = (const float*)d_in[9];
  const float* b5 = (const float*)d_in[10];
  const float* w6 = (const float*)d_in[11];
  const float* b6 = (const float*)d_in[12];
  float* out = (float*)d_out;
  const int B = in_sizes[0] / (3 * 1024 * 1024);

  // carve workspace
  char* ws = (char*)d_ws;
  size_t off = 0;
  auto carve = [&](size_t bytes) -> void* {
    void* p = (void*)(ws + off);
    off = (off + bytes + 255) & ~(size_t)255;
    return p;
  };
  _Float16* act0 = (_Float16*)carve((size_t)B * 65536u * 4u * 2u);
  _Float16* bufA = (_Float16*)carve((size_t)B * 65536u * 96u * 2u);
  _Float16* bufB = (_Float16*)carve((size_t)B * 65536u * 96u * 2u);
  _Float16* wp2 = (_Float16*)carve((size_t)32 * 160 * 2);
  _Float16* wp3 = (_Float16*)carve((size_t)32 * 288 * 2);
  _Float16* wp4 = (_Float16*)carve((size_t)64 * 288 * 2);
  _Float16* wp5 = (_Float16*)carve((size_t)64 * 576 * 2);
  _Float16* wp6 = (_Float16*)carve((size_t)96 * 576 * 2);
  float* gridbuf = (float*)carve((size_t)B * 12u * 2048u * 4u);

  const int TB = 256;
  // weight prep (tiny)
  wprep_kernel<<<(32 * 160 + TB - 1) / TB, TB, 0, stream>>>(w2, wp2, 16, 160, 32 * 160);
  wprep_kernel<<<(32 * 288 + TB - 1) / TB, TB, 0, stream>>>(w3, wp3, 32, 288, 32 * 288);
  wprep_kernel<<<(64 * 288 + TB - 1) / TB, TB, 0, stream>>>(w4, wp4, 32, 288, 64 * 288);
  wprep_kernel<<<(64 * 576 + TB - 1) / TB, TB, 0, stream>>>(w5, wp5, 64, 576, 64 * 576);
  wprep_kernel<<<(96 * 576 + TB - 1) / TB, TB, 0, stream>>>(w6, wp6, 64, 576, 96 * 576);

  // downsample + conv1
  int nLow = B * 65536;
  downsample_kernel<<<(nLow + TB - 1) / TB, TB, 0, stream>>>(x, act0, nLow);
  conv1_kernel<<<(nLow + TB - 1) / TB, TB, 0, stream>>>(act0, w1, b1, bufA, nLow);

  // WMMA conv stack (ping-pong A<->B)
  dim3 cgrid(65536 / 32, 1, B);
  conv_wmma_kernel<16, 32, 160, true><<<cgrid, TB, 0, stream>>>(bufA, wp2, b2, bufB);
  conv_wmma_kernel<32, 32, 288, true><<<cgrid, TB, 0, stream>>>(bufB, wp3, b3, bufA);
  conv_wmma_kernel<32, 64, 288, true><<<cgrid, TB, 0, stream>>>(bufA, wp4, b4, bufB);
  conv_wmma_kernel<64, 64, 576, true><<<cgrid, TB, 0, stream>>>(bufB, wp5, b5, bufA);
  conv_wmma_kernel<64, 96, 576, false><<<cgrid, TB, 0, stream>>>(bufA, wp6, b6, bufB);

  // bilateral grid + fused slice/apply
  int nGrid = B * 96 * 256;
  grid_build_kernel<<<(nGrid + TB - 1) / TB, TB, 0, stream>>>(bufB, gridbuf, nGrid);
  int nPix = B * 1048576;
  slice_apply_kernel<<<(nPix + TB - 1) / TB, TB, 0, stream>>>(x, gridbuf, out, nPix);
}
